// DSKR_41016937676897
// MI455X (gfx1250) — matrix operand order repaired and checked by reading, not compiled, not run on hardware
//
#include <hip/hip_runtime.h>

// ---------------------------------------------------------------------------
// GNN forward on MI455X (gfx1250): all dense layers via v_wmma_f32_16x16x32_bf16
// (f32 accumulate). Weights pre-packed into WMMA B-fragment order so A and B
// both load as contiguous global_load_b128 pairs; each wave computes two
// independent 16x16 accumulator chains to keep the matrix pipe busy.
// ---------------------------------------------------------------------------

#define B_     2
#define NC_    4096
#define NT_    1024
#define NPG_   (NC_ + NT_)       // 5120 nodes per graph
#define NG_    (B_ * NPG_)       // 10240 nodes total
#define KNN_   10
#define EE_    (NG_ * KNN_)      // 102400 edges
#define NTEST_ (B_ * NT_)        // 2048 test nodes

typedef __attribute__((ext_vector_type(16))) __bf16 v16bf;
typedef __attribute__((ext_vector_type(8)))  float  v8f;

__device__ __forceinline__ __bf16 f2bf(float f) {
  unsigned u = __builtin_bit_cast(unsigned, f);
  unsigned r = u + 0x7FFFu + ((u >> 16) & 1u);      // round-to-nearest-even
  unsigned short h = (unsigned short)(r >> 16);
  return __builtin_bit_cast(__bf16, h);
}
__device__ __forceinline__ float bf2f(__bf16 b) {
  unsigned u = ((unsigned)__builtin_bit_cast(unsigned short, b)) << 16;
  return __builtin_bit_cast(float, u);
}
__device__ __forceinline__ float gelu_tanh(float x) {
  float x3 = x * x * x;
  return 0.5f * x * (1.0f + tanhf(0.7978845608028654f * (x + 0.044715f * x3)));
}

// ---------------- weight repack: f32 [K x N] -> bf16 WMMA-B-fragment order.
// Fragment region (tn, kb): 32 lanes x 16 contiguous bf16 per lane.
// lane element e:  K = kb*32 + 16*(lane/16) + e,  N = tn*16 + (lane%16)
__global__ __launch_bounds__(256) void k_pack_w_frag(const float* __restrict__ W,
                                                     __bf16* __restrict__ Wp,
                                                     int K, int N, int Kpad, int Npad) {
  int idx = blockIdx.x * 256 + threadIdx.x;
  if (idx >= Kpad * Npad) return;
  int e    = idx & 15;
  int lane = (idx >> 4) & 31;
  int KB   = Kpad >> 5;
  int kb   = (idx >> 9) % KB;
  int tn   = idx / (KB << 9);
  int k = kb * 32 + ((lane >> 4) << 4) + e;
  int n = (tn << 4) + (lane & 15);
  float v = (k < K && n < N) ? W[(size_t)k * N + n] : 0.0f;
  Wp[idx] = f2bf(v);
}

// ---------------- generic WMMA GEMM: Y = act(A[MxKPAD] * W + b)
// NTL adjacent 16x16 output tiles per wave32 -> NTL independent WMMA chains,
// A fragment reused across tiles. K-loop compile-time unrolled.
template <int KPAD, int NTL>
__global__ __launch_bounds__(256) void k_gemm_bias_act(
    const __bf16* __restrict__ A, const __bf16* __restrict__ Wp,
    const float* __restrict__ bias,
    __bf16* __restrict__ Ybf, float* __restrict__ Yf,
    int M, int Npad, int Norig, int act) {
  constexpr int KB = KPAD / 32;
  int wid  = (blockIdx.x * 256 + threadIdx.x) >> 5;
  int lane = threadIdx.x & 31;
  int groupsN = (Npad >> 4) / NTL;
  if (wid >= (M >> 4) * groupsN) return;
  int tm = wid / groupsN, tn0 = (wid - tm * groupsN) * NTL;
  int khi = lane >> 4;             // 0: lanes 0-15, 1: lanes 16-31
  int lr  = lane & 15;
  const uint4* arow  = (const uint4*)(A + (size_t)((tm << 4) + lr) * KPAD);
  const uint4* bfrag = (const uint4*)Wp + ((size_t)tn0 * KB * 32 + lane) * 2;
  v8f acc[NTL];
#pragma unroll
  for (int t = 0; t < NTL; ++t) acc[t] = (v8f){};
#pragma unroll
  for (int kb = 0; kb < KB; ++kb) {
    union { v16bf v; uint4 q[2]; } au;
    au.q[0] = arow[kb * 4 + 2 * khi];      // K = kb*32 + 16*khi     .. +7
    au.q[1] = arow[kb * 4 + 2 * khi + 1];  // K = kb*32 + 16*khi + 8 .. +15
#pragma unroll
    for (int t = 0; t < NTL; ++t) {
      union { v16bf v; uint4 q[2]; } bu;
      bu.q[0] = bfrag[(size_t)t * KB * 64 + kb * 64];     // 32B contiguous/lane
      bu.q[1] = bfrag[(size_t)t * KB * 64 + kb * 64 + 1];
      acc[t] = __builtin_amdgcn_wmma_f32_16x16x32_bf16(false, au.v, false, bu.v,
                                                       (short)0, acc[t], false, false);
    }
  }
#pragma unroll
  for (int t = 0; t < NTL; ++t) {
    int n = ((tn0 + t) << 4) + lr;
#pragma unroll
    for (int r = 0; r < 8; ++r) {
      int m = (tm << 4) + r + (khi << 3);  // C: vgpr r -> M=r (lo) / 8+r (hi)
      float v = acc[t][r];
      if (n < Norig) {
        if (bias) v += bias[n];
        if (act)  v = gelu_tanh(v);
      } else {
        v = 0.0f;                           // padded col stays exactly zero
      }
      size_t o = (size_t)m * Npad + n;
      if (Ybf) Ybf[o] = f2bf(v);
      if (Yf)  Yf[o]  = v;
    }
  }
}

// ---------------- kNN: one thread per receiver, insertion top-10 (stable ties)
__global__ __launch_bounds__(256) void k_knn(const float* __restrict__ s_ctx,
                                             const float* __restrict__ s_test,
                                             int* __restrict__ senders,
                                             float* __restrict__ dists) {
  int g = blockIdx.x * 256 + threadIdx.x;
  if (g >= NG_) return;
  int b = g / NPG_, r = g - b * NPG_;
  const float* rx = (r < NC_) ? (s_ctx + ((size_t)b * NC_ + r) * 3)
                              : (s_test + ((size_t)b * NT_ + (r - NC_)) * 3);
  float px = rx[0], py = rx[1], pz = rx[2];
  float bd[KNN_]; int bi[KNN_];
#pragma unroll
  for (int k = 0; k < KNN_; ++k) { bd[k] = 3.4e38f; bi[k] = 0; }
  const float* tx = s_ctx + (size_t)b * NC_ * 3;
  for (int j = 0; j < NC_; ++j) {
    float dx = px - tx[3 * j], dy = py - tx[3 * j + 1], dz = pz - tx[3 * j + 2];
    float d = sqrtf(dx * dx + dy * dy + dz * dz + 1e-12f);
    if (d < bd[KNN_ - 1]) {
      int p = KNN_ - 1;
      while (p > 0 && bd[p - 1] > d) { bd[p] = bd[p - 1]; bi[p] = bi[p - 1]; --p; }
      bd[p] = d; bi[p] = j;
    }
  }
#pragma unroll
  for (int k = 0; k < KNN_; ++k) {
    senders[(size_t)g * KNN_ + k] = b * NPG_ + bi[k];
    dists[(size_t)g * KNN_ + k]  = bd[k];
  }
}

// ---------------- embed input: [obs_embed(4) | pos(3) | f(1) | zeros..32]
__global__ __launch_bounds__(256) void k_build_x0(const float* __restrict__ s_ctx,
                                                  const float* __restrict__ f_ctx,
                                                  const float* __restrict__ s_test,
                                                  const float* __restrict__ obs_e,
                                                  __bf16* __restrict__ X0) {
  int g = blockIdx.x * 256 + threadIdx.x;
  if (g >= NG_) return;
  int b = g / NPG_, r = g - b * NPG_;
  float f[8];
  if (r < NC_) {
#pragma unroll
    for (int j = 0; j < 4; ++j) f[j] = obs_e[4 + j];          // obs=1
    const float* s = s_ctx + ((size_t)b * NC_ + r) * 3;
    f[4] = s[0]; f[5] = s[1]; f[6] = s[2];
    f[7] = f_ctx[(size_t)b * NC_ + r];
  } else {
#pragma unroll
    for (int j = 0; j < 4; ++j) f[j] = obs_e[j];              // obs=0
    const float* s = s_test + ((size_t)b * NT_ + (r - NC_)) * 3;
    f[4] = s[0]; f[5] = s[1]; f[6] = s[2];
    f[7] = 0.0f;
  }
  __bf16* row = X0 + (size_t)g * 32;
#pragma unroll
  for (int j = 0; j < 8; ++j)  row[j] = f2bf(f[j]);
#pragma unroll
  for (int j = 8; j < 32; ++j) row[j] = __builtin_bit_cast(__bf16, (unsigned short)0);
}

// ---------------- LayerNorm over 64 feats, wave per row; optional residual
__global__ __launch_bounds__(256) void k_ln(const float* __restrict__ X,
                                            const float* __restrict__ RES,
                                            const float* __restrict__ gg,
                                            const float* __restrict__ bb,
                                            float* __restrict__ outF,
                                            __bf16* __restrict__ outB, int M) {
  int w = (blockIdx.x * 256 + threadIdx.x) >> 5;
  int lane = threadIdx.x & 31;
  if (w >= M) return;
  const float* row = X + (size_t)w * 64;
  float v0 = row[lane], v1 = row[lane + 32];
  if (RES) { const float* rr = RES + (size_t)w * 64; v0 += rr[lane]; v1 += rr[lane + 32]; }
  float s = v0 + v1;
#pragma unroll
  for (int o = 16; o > 0; o >>= 1) s += __shfl_xor(s, o, 32);
  float mu = s * (1.0f / 64.0f);
  float d0 = v0 - mu, d1 = v1 - mu;
  float q = d0 * d0 + d1 * d1;
#pragma unroll
  for (int o = 16; o > 0; o >>= 1) q += __shfl_xor(q, o, 32);
  float inv = rsqrtf(q * (1.0f / 64.0f) + 1e-6f);
  float o0 = d0 * inv * gg[lane]      + bb[lane];
  float o1 = d1 * inv * gg[lane + 32] + bb[lane + 32];
  if (outF) { float* p = outF + (size_t)w * 64; p[lane] = o0; p[lane + 32] = o1; }
  if (outB) { __bf16* p = outB + (size_t)w * 64; p[lane] = f2bf(o0); p[lane + 32] = f2bf(o1); }
}

// ---------------- edge MLP input: [x_s(64) | x_r(64) | dist(1)+pad  or  m_prev(64)]
__global__ __launch_bounds__(256) void k_build_edge_in(const __bf16* __restrict__ nodesB,
                                                       const int* __restrict__ senders,
                                                       const float* __restrict__ dist,
                                                       const __bf16* __restrict__ mPrev,
                                                       __bf16* __restrict__ EI, int Kpad) {
  int e = (blockIdx.x * 256 + threadIdx.x) >> 5;
  int lane = threadIdx.x & 31;
  if (e >= EE_) return;
  int s = senders[e];
  int r = e / KNN_;                      // receiver = repeat(arange(B*N), K)
  __bf16* row = EI + (size_t)e * Kpad;
  const __bf16* xs = nodesB + (size_t)s * 64;
  const __bf16* xr = nodesB + (size_t)r * 64;
  row[lane]           = xs[lane];
  row[lane + 32]      = xs[lane + 32];
  row[64 + lane]      = xr[lane];
  row[64 + lane + 32] = xr[lane + 32];
  if (mPrev) {
    const __bf16* mp = mPrev + (size_t)e * 64;
    row[128 + lane]      = mp[lane];
    row[128 + lane + 32] = mp[lane + 32];
  } else {                                // block 0: Kpad == 160
    if (lane == 0) row[128] = f2bf(dist[e]);
    else           row[128 + lane] = __builtin_bit_cast(__bf16, (unsigned short)0);
  }
}

// ---------------- per-receiver mean over K edges + node MLP input concat
__global__ __launch_bounds__(256) void k_build_node_in(const float* __restrict__ nodesF,
                                                       const __bf16* __restrict__ mBf,
                                                       __bf16* __restrict__ NI) {
  int g = (blockIdx.x * 256 + threadIdx.x) >> 5;
  int lane = threadIdx.x & 31;
  if (g >= NG_) return;
  const __bf16* base = mBf + (size_t)g * KNN_ * 64;
  float a0 = 0.0f, a1 = 0.0f;
#pragma unroll
  for (int k = 0; k < KNN_; ++k) {
    a0 += bf2f(base[k * 64 + lane]);
    a1 += bf2f(base[k * 64 + lane + 32]);
  }
  a0 *= (1.0f / KNN_); a1 *= (1.0f / KNN_);
  __bf16* row = NI + (size_t)g * 128;
  const float* nf = nodesF + (size_t)g * 64;
  row[lane]           = f2bf(nf[lane]);
  row[lane + 32]      = f2bf(nf[lane + 32]);
  row[64 + lane]      = f2bf(a0);
  row[64 + lane + 32] = f2bf(a1);
}

// ---------------- head input: test-node slice of nodes, f32 -> bf16
__global__ __launch_bounds__(256) void k_build_head_in(const float* __restrict__ nodesF,
                                                       __bf16* __restrict__ XH) {
  int t = (blockIdx.x * 256 + threadIdx.x) >> 5;
  int lane = threadIdx.x & 31;
  if (t >= NTEST_) return;
  int b = t / NT_, rt = t - b * NT_;
  const float* nf = nodesF + ((size_t)(b * NPG_ + NC_ + rt)) * 64;
  __bf16* row = XH + (size_t)t * 64;
  row[lane]      = f2bf(nf[lane]);
  row[lane + 32] = f2bf(nf[lane + 32]);
}

// ---------------- split [mu, logvar] -> out = [mu..., exp(logvar/2)...]
__global__ __launch_bounds__(256) void k_finalize(const float* __restrict__ HO,
                                                  float* __restrict__ out) {
  int t = blockIdx.x * 256 + threadIdx.x;
  if (t >= NTEST_) return;
  out[t]          = HO[(size_t)t * 16 + 0];
  out[NTEST_ + t] = expf(HO[(size_t)t * 16 + 1] * 0.5f);
}

// ---------------------------------------------------------------------------
extern "C" void kernel_launch(void* const* d_in, const int* in_sizes, int n_in,
                              void* d_out, int out_size, void* d_ws, size_t ws_size,
                              hipStream_t stream) {
  (void)in_sizes; (void)n_in; (void)out_size; (void)ws_size;
  const float* s_ctx  = (const float*)d_in[0];
  const float* f_ctx  = (const float*)d_in[1];
  const float* s_test = (const float*)d_in[2];
  // params flattened as a JAX pytree (dict keys sorted; lists in order):
  // blocks[6 x {edge:[{W,b}x2], ln_b, ln_g, node:[{W,b}x2]}] (3..62),
  // embed_all (63..68), embed_obs (69), head (70..75), norm_b (76), norm_g (77)
  auto F = [&](int i) { return (const float*)d_in[i]; };
  const int IB = 3, IEMB = 63, IOBS = 69, IHEAD = 70, INORMB = 76, INORMG = 77;

  char* wsp = (char*)d_ws;
  size_t off = 0;
  auto alloc = [&](size_t bytes) -> void* {
    void* p = wsp + off;
    off += (bytes + 255) & ~(size_t)255;
    return p;
  };

  // packed bf16 weights (WMMA B-fragment order)
  __bf16* eWp0 = (__bf16*)alloc((size_t)32 * 256 * 2);
  __bf16* eWp1 = (__bf16*)alloc((size_t)256 * 128 * 2);
  __bf16* eWp2 = (__bf16*)alloc((size_t)128 * 64 * 2);
  __bf16* hWp0 = (__bf16*)alloc((size_t)64 * 256 * 2);
  __bf16* hWp1 = (__bf16*)alloc((size_t)256 * 64 * 2);
  __bf16* hWp2 = (__bf16*)alloc((size_t)64 * 16 * 2);
  __bf16 *beW0[6], *beW1[6], *bnW0[6], *bnW1[6];
  int kpadE[6];
  for (int j = 0; j < 6; ++j) {
    kpadE[j] = (j == 0) ? 160 : 192;                 // 129 -> 160 pad, else 192
    beW0[j] = (__bf16*)alloc((size_t)kpadE[j] * 128 * 2);
    beW1[j] = (__bf16*)alloc((size_t)128 * 64 * 2);
    bnW0[j] = (__bf16*)alloc((size_t)128 * 128 * 2);
    bnW1[j] = (__bf16*)alloc((size_t)128 * 64 * 2);
  }

  // activations
  __bf16* X0     = (__bf16*)alloc((size_t)NG_ * 32 * 2);
  __bf16* Hb1    = (__bf16*)alloc((size_t)NG_ * 256 * 2);
  __bf16* Hb2    = (__bf16*)alloc((size_t)NG_ * 128 * 2);
  float*  EmbF   = (float*) alloc((size_t)NG_ * 64 * 4);
  float*  nodesF = (float*) alloc((size_t)NG_ * 64 * 4);
  __bf16* nodesB = (__bf16*)alloc((size_t)NG_ * 64 * 2);
  int*    senders = (int*)  alloc((size_t)EE_ * 4);
  float*  dist    = (float*)alloc((size_t)EE_ * 4);
  __bf16* EI  = (__bf16*)alloc((size_t)EE_ * 192 * 2);
  __bf16* EH  = (__bf16*)alloc((size_t)EE_ * 128 * 2);
  __bf16* Mbf = (__bf16*)alloc((size_t)EE_ * 64 * 2);
  __bf16* NI  = (__bf16*)alloc((size_t)NG_ * 128 * 2);
  __bf16* NH  = (__bf16*)alloc((size_t)NG_ * 128 * 2);
  float*  UPD = (float*) alloc((size_t)NG_ * 64 * 4);
  __bf16* XH  = (__bf16*)alloc((size_t)NTEST_ * 64 * 2);
  __bf16* HH1 = (__bf16*)alloc((size_t)NTEST_ * 256 * 2);
  __bf16* HH2 = (__bf16*)alloc((size_t)NTEST_ * 64 * 2);
  float*  HO  = (float*) alloc((size_t)NTEST_ * 16 * 4);

  auto pack = [&](int idx, int K, int N, int Kpad, int Npad, __bf16* dst) {
    int tot = Kpad * Npad;
    k_pack_w_frag<<<(tot + 255) / 256, 256, 0, stream>>>(F(idx), dst, K, N, Kpad, Npad);
  };
  pack(IEMB + 0, 8, 256, 32, 256, eWp0);
  pack(IEMB + 2, 256, 128, 256, 128, eWp1);
  pack(IEMB + 4, 128, 64, 128, 64, eWp2);
  pack(IHEAD + 0, 64, 256, 64, 256, hWp0);
  pack(IHEAD + 2, 256, 64, 256, 64, hWp1);
  pack(IHEAD + 4, 64, 2, 64, 16, hWp2);
  for (int j = 0; j < 6; ++j) {
    int base = IB + j * 10;
    int ein = (j == 0) ? 129 : 192;
    pack(base + 0, ein, 128, kpadE[j], 128, beW0[j]);
    pack(base + 2, 128, 64, 128, 64, beW1[j]);
    pack(base + 6, 128, 128, 128, 128, bnW0[j]);
    pack(base + 8, 128, 64, 128, 64, bnW1[j]);
  }

  auto gemm = [&](const __bf16* A, const __bf16* W, const float* bias,
                  __bf16* Ybf, float* Yf, int M, int Kpad, int Npad, int Norig, int act) {
    int ntl = (Npad >= 32) ? 2 : 1;
    int waves = (M >> 4) * ((Npad >> 4) / ntl);
    int blocks = (waves + 7) / 8;                     // 8 waves / 256-thread block
    if (ntl == 2) {
      switch (Kpad) {
        case 32:  k_gemm_bias_act<32, 2><<<blocks, 256, 0, stream>>>(A, W, bias, Ybf, Yf, M, Npad, Norig, act); break;
        case 64:  k_gemm_bias_act<64, 2><<<blocks, 256, 0, stream>>>(A, W, bias, Ybf, Yf, M, Npad, Norig, act); break;
        case 128: k_gemm_bias_act<128, 2><<<blocks, 256, 0, stream>>>(A, W, bias, Ybf, Yf, M, Npad, Norig, act); break;
        case 160: k_gemm_bias_act<160, 2><<<blocks, 256, 0, stream>>>(A, W, bias, Ybf, Yf, M, Npad, Norig, act); break;
        case 192: k_gemm_bias_act<192, 2><<<blocks, 256, 0, stream>>>(A, W, bias, Ybf, Yf, M, Npad, Norig, act); break;
        case 256: k_gemm_bias_act<256, 2><<<blocks, 256, 0, stream>>>(A, W, bias, Ybf, Yf, M, Npad, Norig, act); break;
        default: break;
      }
    } else {
      switch (Kpad) {
        case 64:  k_gemm_bias_act<64, 1><<<blocks, 256, 0, stream>>>(A, W, bias, Ybf, Yf, M, Npad, Norig, act); break;
        default: break;
      }
    }
  };

  // embed MLP + LayerNorm
  k_build_x0<<<(NG_ + 255) / 256, 256, 0, stream>>>(s_ctx, f_ctx, s_test, F(IOBS), X0);
  gemm(X0,  eWp0, F(IEMB + 1), Hb1, nullptr, NG_, 32, 256, 256, 1);
  gemm(Hb1, eWp1, F(IEMB + 3), Hb2, nullptr, NG_, 256, 128, 128, 1);
  gemm(Hb2, eWp2, F(IEMB + 5), nullptr, EmbF, NG_, 128, 64, 64, 0);
  k_ln<<<NG_ / 8, 256, 0, stream>>>(EmbF, nullptr, F(INORMG), F(INORMB), nodesF, nodesB, NG_);

  // kNN graph build
  k_knn<<<(NG_ + 255) / 256, 256, 0, stream>>>(s_ctx, s_test, senders, dist);

  // 6 message-passing blocks
  for (int j = 0; j < 6; ++j) {
    int base = IB + j * 10;
    k_build_edge_in<<<(EE_ * 32) / 256, 256, 0, stream>>>(
        nodesB, senders, dist, (j == 0) ? nullptr : Mbf, EI, kpadE[j]);
    gemm(EI, beW0[j], F(base + 1), EH, nullptr, EE_, kpadE[j], 128, 128, 1);
    gemm(EH, beW1[j], F(base + 3), Mbf, nullptr, EE_, 128, 64, 64, 0);
    k_build_node_in<<<(NG_ * 32) / 256, 256, 0, stream>>>(nodesF, Mbf, NI);
    gemm(NI, bnW0[j], F(base + 7), NH, nullptr, NG_, 128, 128, 128, 1);
    gemm(NH, bnW1[j], F(base + 9), nullptr, UPD, NG_, 128, 64, 64, 0);
    k_ln<<<NG_ / 8, 256, 0, stream>>>(nodesF, UPD, F(base + 5), F(base + 4),
                                      nodesF, nodesB, NG_);
  }

  // head MLP + output split
  k_build_head_in<<<(NTEST_ * 32) / 256, 256, 0, stream>>>(nodesF, XH);
  gemm(XH,  hWp0, F(IHEAD + 1), HH1, nullptr, NTEST_, 64, 256, 256, 1);
  gemm(HH1, hWp1, F(IHEAD + 3), HH2, nullptr, NTEST_, 256, 64, 64, 1);
  gemm(HH2, hWp2, F(IHEAD + 5), nullptr, HO, NTEST_, 64, 16, 2, 0);
  k_finalize<<<(NTEST_ + 255) / 256, 256, 0, stream>>>(HO, (float*)d_out);
}
